// NeRDPixelDecoder_73924977098933
// MI455X (gfx1250) — matrix operand order repaired and checked by reading, not compiled, not run on hardware
//
#include <hip/hip_runtime.h>

// ---------------------------------------------------------------------------
// SIREN pixel decoder for MI455X (gfx1250), wave32 + WMMA f16 (f32 accumulate)
// Layer0: [64px, 3202] @ [3202,256]; Layers1/2: [64,256]@[256,256]; L3 scalar.
// ---------------------------------------------------------------------------

typedef __attribute__((ext_vector_type(16))) _Float16 v16h;
typedef __attribute__((ext_vector_type(8)))  float    v8f;

union Frag {
  v16h  v;
  uint4 q[2];
};

#define K0PAD   3232      // 3202 padded up to multiple of 32
#define ASTRIDE 424       // LDS A row stride in halves (212 dwords: conflict-free b128)
#define HSTRIDE 264       // LDS h row stride in halves (132 dwords: conflict-free b128)
#define LDS_BYTES (64 * ASTRIDE * 2)   // 54272 B; also covers 64*HSTRIDE*2 = 33792 B

// ---- weight prep: f32 [K,256] -> f16 transposed [256, Kpad] ----------------
__global__ void prep_w0_kernel(const float* __restrict__ W0,
                               _Float16* __restrict__ Wt0) {
  int idx = blockIdx.x * 256 + threadIdx.x;   // 256*3232 threads exactly
  int n = idx / K0PAD;
  int k = idx - n * K0PAD;
  float v = (k < 3202) ? W0[k * 256 + n] : 0.0f;
  Wt0[n * K0PAD + k] = (_Float16)v;
}

__global__ void prep_w256_kernel(const float* __restrict__ Win,
                                 _Float16* __restrict__ Wt) {
  int idx = blockIdx.x * 256 + threadIdx.x;   // 65536 threads exactly
  int n = idx >> 8;
  int k = idx & 255;
  Wt[n * 256 + k] = (_Float16)Win[k * 256 + n];
}

__device__ __forceinline__ v8f wmma_f16(const Frag& a, const Frag& b, v8f c) {
  // (neg_a, A, neg_b, B, c_mod, C, reuse_a, reuse_b)
  return __builtin_amdgcn_wmma_f32_16x16x32_f16(false, a.v, false, b.v,
                                                (short)0, c, false, false);
}

__device__ __forceinline__ v8f vzero8() {
  v8f z;
#pragma unroll
  for (int i = 0; i < 8; ++i) z[i] = 0.0f;
  return z;
}

// ---- main fused decoder ----------------------------------------------------
__global__ __launch_bounds__(256) void siren_decode_kernel(
    const float* __restrict__ xi,
    const _Float16* __restrict__ Wt0, const float* __restrict__ b0,
    const _Float16* __restrict__ Wt1, const float* __restrict__ b1,
    const _Float16* __restrict__ Wt2, const float* __restrict__ b2,
    const float* __restrict__ W3, const float* __restrict__ b3,
    float* __restrict__ out)
{
  __shared__ __align__(16) unsigned char smem[LDS_BYTES];
  _Float16* Abuf = (_Float16*)smem;   // [64][ASTRIDE], layer-0 staging
  _Float16* hbuf = (_Float16*)smem;   // [64][HSTRIDE], aliases Abuf (disjoint lifetime)

  const int tid  = threadIdx.x;
  const int lane = tid & 31;
  const int wave = tid >> 5;          // 8 waves, wave w owns N cols [32w, 32w+32)
  const int g    = lane >> 4;         // half-wave group
  const int r    = lane & 15;

  // 512 tiles: 256 per batch image, 2 tiles (of 64 px) per image row
  const int t   = blockIdx.x;
  const int bb  = t >> 8;
  const int rem = t & 255;
  const int y   = rem >> 1;
  const int x0  = (rem & 1) << 6;

  const float* xib = xi + (bb << 21);  // + b*128*128*128

  v8f acc[4][2];
#pragma unroll
  for (int mt = 0; mt < 4; ++mt) {
    acc[mt][0] = vzero8();
    acc[mt][1] = vzero8();
  }

  const _Float16* w0row0 = Wt0 + (wave * 32 + r) * K0PAD;       // N-tile 0 col
  const _Float16* w0row1 = w0row0 + 16 * K0PAD;                 // N-tile 1 col

  // ======== layer 0: 8 super-chunks of 16 channels (K=400 each, LDS-staged) =
  for (int s = 0; s < 8; ++s) {
    __syncthreads();
    // gather 5x5 patches (SAME zero pad) for 64 pixels x 16 channels -> f16
    for (int i = tid; i < 64 * 416; i += 256) {   // 104 uniform iterations
      int m  = i / 416;
      int kk = i - m * 416;
      float v = 0.0f;
      if (kk < 400) {
        int c  = (s << 4) + kk / 25;
        int p  = kk - (kk / 25) * 25;             // dy*5+dx
        int yy = y + p / 5 - 2;
        int xx = x0 + m + (p - (p / 5) * 5) - 2;
        if ((unsigned)yy < 128u && (unsigned)xx < 128u)
          v = xib[(c << 14) + (yy << 7) + xx];
      }
      Abuf[m * ASTRIDE + kk] = (_Float16)v;
    }
    __syncthreads();

    const int kbase = s * 400;
    for (int kc = 0; kc < 13; ++kc) {   // 13*32=416; tail zeros in A kill overlap terms
      const int kg = kbase + kc * 32 + 16 * g;
      Frag fb0, fb1;
      fb0.q[0] = *(const uint4*)(w0row0 + kg);
      fb0.q[1] = *(const uint4*)(w0row0 + kg + 8);
      fb1.q[0] = *(const uint4*)(w0row1 + kg);
      fb1.q[1] = *(const uint4*)(w0row1 + kg + 8);
#pragma unroll
      for (int mt = 0; mt < 4; ++mt) {
        Frag fa;
        const _Float16* ap = Abuf + (mt * 16 + r) * ASTRIDE + kc * 32 + 16 * g;
        fa.q[0] = *(const uint4*)(ap);
        fa.q[1] = *(const uint4*)(ap + 8);
        acc[mt][0] = wmma_f16(fa, fb0, acc[mt][0]);
        acc[mt][1] = wmma_f16(fa, fb1, acc[mt][1]);
      }
    }
  }

  // ======== layer 0: coordinate rows K=3200 (gx), 3201 (gy) ================
  {
    const int kg = 3200 + 16 * g;
    Frag fb0, fb1;
    fb0.q[0] = *(const uint4*)(w0row0 + kg);
    fb0.q[1] = *(const uint4*)(w0row0 + kg + 8);
    fb1.q[0] = *(const uint4*)(w0row1 + kg);
    fb1.q[1] = *(const uint4*)(w0row1 + kg + 8);
    const float gyv = -1.0f + (2.0f / 127.0f) * (float)y;
#pragma unroll
    for (int mt = 0; mt < 4; ++mt) {
      Frag fa;
      fa.q[0] = uint4{0u, 0u, 0u, 0u};
      fa.q[1] = uint4{0u, 0u, 0u, 0u};
      if (g == 0) {   // rel K 0,1 live only in half-wave group 0 elements 0,1
        float gxv = -1.0f + (2.0f / 127.0f) * (float)(x0 + mt * 16 + r);
        fa.v[0] = (_Float16)gxv;
        fa.v[1] = (_Float16)gyv;
      }
      acc[mt][0] = wmma_f16(fa, fb0, acc[mt][0]);
      acc[mt][1] = wmma_f16(fa, fb1, acc[mt][1]);
    }
  }

  // ======== epilogue 0: h0 = sin(30*(acc+b0)) -> hbuf (Abuf now dead) ======
  __syncthreads();
#pragma unroll
  for (int mt = 0; mt < 4; ++mt)
#pragma unroll
    for (int j = 0; j < 2; ++j) {
      const int n = wave * 32 + j * 16 + r;
      const float bia = b0[n];
#pragma unroll
      for (int v = 0; v < 8; ++v) {
        const int m = mt * 16 + v + 8 * g;   // C-tile layout: M = vgpr + 8*(lane/16)
        hbuf[m * HSTRIDE + n] = (_Float16)__sinf(30.0f * (acc[mt][j][v] + bia));
      }
    }
  __syncthreads();

  // ======== layers 1 and 2: [64,256] @ [256,256], K = 8 chunks of 32 =======
  for (int layer = 0; layer < 2; ++layer) {
    const _Float16* Wt = layer ? Wt2 : Wt1;
    const float*    bi = layer ? b2  : b1;
#pragma unroll
    for (int mt = 0; mt < 4; ++mt) {
      acc[mt][0] = vzero8();
      acc[mt][1] = vzero8();
    }
    const _Float16* wr0 = Wt + (wave * 32 + r) * 256;
    const _Float16* wr1 = wr0 + 16 * 256;
#pragma unroll
    for (int kc = 0; kc < 8; ++kc) {
      const int kg = kc * 32 + 16 * g;
      Frag fb0, fb1;
      fb0.q[0] = *(const uint4*)(wr0 + kg);
      fb0.q[1] = *(const uint4*)(wr0 + kg + 8);
      fb1.q[0] = *(const uint4*)(wr1 + kg);
      fb1.q[1] = *(const uint4*)(wr1 + kg + 8);
#pragma unroll
      for (int mt = 0; mt < 4; ++mt) {
        Frag fa;
        const _Float16* ap = hbuf + (mt * 16 + r) * HSTRIDE + kc * 32 + 16 * g;
        fa.q[0] = *(const uint4*)(ap);
        fa.q[1] = *(const uint4*)(ap + 8);
        acc[mt][0] = wmma_f16(fa, fb0, acc[mt][0]);
        acc[mt][1] = wmma_f16(fa, fb1, acc[mt][1]);
      }
    }
    __syncthreads();   // all waves finished reading h_prev; safe to overwrite
#pragma unroll
    for (int mt = 0; mt < 4; ++mt)
#pragma unroll
      for (int j = 0; j < 2; ++j) {
        const int n = wave * 32 + j * 16 + r;
        const float bia = bi[n];
#pragma unroll
        for (int v = 0; v < 8; ++v) {
          const int m = mt * 16 + v + 8 * g;
          hbuf[m * HSTRIDE + n] = (_Float16)__sinf(30.0f * (acc[mt][j][v] + bia));
        }
      }
    __syncthreads();
  }

  // ======== layer 3: [64,256] @ [256,3] + b3, scalar f32 ===================
  if (tid < 192) {
    const int m = tid / 3;
    const int o = tid - m * 3;
    float sacc = b3[o];
    const _Float16* hp = hbuf + m * HSTRIDE;
    for (int k = 0; k < 256; ++k)
      sacc += (float)hp[k] * W3[k * 3 + o];
    out[((bb * 3 + o) << 14) + (y << 7) + x0 + m] = sacc;
  }
}

// ---------------------------------------------------------------------------
extern "C" void kernel_launch(void* const* d_in, const int* in_sizes, int n_in,
                              void* d_out, int out_size, void* d_ws, size_t ws_size,
                              hipStream_t stream) {
  (void)in_sizes; (void)n_in; (void)out_size; (void)ws_size;
  const float* xi = (const float*)d_in[0];
  const float* W0 = (const float*)d_in[1];
  const float* b0 = (const float*)d_in[2];
  const float* W1 = (const float*)d_in[3];
  const float* b1 = (const float*)d_in[4];
  const float* W2 = (const float*)d_in[5];
  const float* b2 = (const float*)d_in[6];
  const float* W3 = (const float*)d_in[7];
  const float* b3 = (const float*)d_in[8];

  char* ws = (char*)d_ws;
  _Float16* Wt0 = (_Float16*)(ws);                                   // 256*3232*2 B
  _Float16* Wt1 = (_Float16*)(ws + (size_t)256 * K0PAD * 2);         // 256*256*2 B
  _Float16* Wt2 = (_Float16*)(ws + (size_t)256 * K0PAD * 2 + 256 * 256 * 2);

  prep_w0_kernel  <<<K0PAD, 256, 0, stream>>>(W0, Wt0);   // 3232 blocks
  prep_w256_kernel<<<256,   256, 0, stream>>>(W1, Wt1);
  prep_w256_kernel<<<256,   256, 0, stream>>>(W2, Wt2);

  siren_decode_kernel<<<512, 256, 0, stream>>>(
      xi, Wt0, b0, Wt1, b1, Wt2, b2, W3, b3, (float*)d_out);
}
